// NHAN_14585708937219
// MI455X (gfx1250) — compile-verified, matrix operand order checked by reading, not run on hardware
//
#include <hip/hip_runtime.h>
#include <hip/hip_bf16.h>
#include <math.h>

typedef __attribute__((ext_vector_type(16))) __bf16 v16bf;
typedef __attribute__((ext_vector_type(8)))  float  v8f;

#define ACT_NONE 0
#define ACT_TANH 1
#define ACT_RELU 2

// native f32 -> bf16 (clang lowers to v_cvt_*bf16* with RNE on gfx1250)
__device__ __forceinline__ unsigned short f2bf(float f) {
  return __builtin_bit_cast(unsigned short, (__bf16)f);
}
__device__ __forceinline__ __bf16 bfc(unsigned short u) {
  return __builtin_bit_cast(__bf16, u);
}
__device__ __forceinline__ unsigned pk2(float lo, float hi) {
  return (unsigned)f2bf(lo) | ((unsigned)f2bf(hi) << 16);
}
__device__ __forceinline__ float sigmoidf_(float x) { return 1.0f / (1.0f + expf(-x)); }

// ---------------------------------------------------------------------------
// Generic WMMA GEMM: C[M,N] = act(A[M,K] @ W[N,K]^T + bias[N])
// Block: 256 threads (8 wave32), tile 64(M) x 64(N), K chunks of 32 (bf16).
// Staging: thread (r = tid>>2, kq = (tid&3)*8) copies 8 contiguous K elems:
//   2x global_load_b128 -> packed bf16 cvt -> one ds_store_b128.
// ---------------------------------------------------------------------------
__global__ __launch_bounds__(256) void k_gemm_bf16(
    const float* __restrict__ A, const float* __restrict__ W,
    const float* __restrict__ bias, float* __restrict__ C,
    int M, int N, int K, int act)
{
  __shared__ unsigned short lA[64][32];
  __shared__ unsigned short lB[64][32];
  const int tid  = threadIdx.x;
  const int lane = tid & 31;
  const int wid  = tid >> 5;
  const int wm   = wid >> 1;              // 0..3 -> 16-row slice
  const int wn   = wid & 1;               // 0..1 -> 32-col slice
  const int m0   = blockIdx.y * 64;
  const int n0   = blockIdx.x * 64;
  const int r    = tid >> 2;              // 0..63 staged row
  const int kq   = (tid & 3) * 8;         // 0,8,16,24 staged k offset

  v8f acc0 = 0.0f, acc1 = 0.0f;

  for (int kb = 0; kb < K; kb += 32) {
    const int gk = kb + kq;
    // ---- stage A tile row slice ----
    {
      const int gm = m0 + r;
      if (gm < M && gk + 8 <= K) {
        const float2* s = (const float2*)(A + (size_t)gm * K + gk);
        float2 f0 = s[0], f1 = s[1], f2 = s[2], f3 = s[3];
        uint4 u;
        u.x = pk2(f0.x, f0.y); u.y = pk2(f1.x, f1.y);
        u.z = pk2(f2.x, f2.y); u.w = pk2(f3.x, f3.y);
        *(uint4*)&lA[r][kq] = u;
      } else {
        const int gmc = gm < M ? gm : (M - 1);
        #pragma unroll
        for (int i = 0; i < 8; ++i) {
          int gki = gk + i;
          int gkc = gki < K ? gki : (K - 1);
          float v = A[(size_t)gmc * K + gkc];
          lA[r][kq + i] = (gm < M && gki < K) ? f2bf(v) : (unsigned short)0;
        }
      }
    }
    // ---- stage B (weight) tile row slice ----
    {
      const int gn = n0 + r;
      if (gn < N && gk + 8 <= K) {
        const float2* s = (const float2*)(W + (size_t)gn * K + gk);
        float2 f0 = s[0], f1 = s[1], f2 = s[2], f3 = s[3];
        uint4 u;
        u.x = pk2(f0.x, f0.y); u.y = pk2(f1.x, f1.y);
        u.z = pk2(f2.x, f2.y); u.w = pk2(f3.x, f3.y);
        *(uint4*)&lB[r][kq] = u;
      } else {
        const int gnc = gn < N ? gn : (N - 1);
        #pragma unroll
        for (int i = 0; i < 8; ++i) {
          int gki = gk + i;
          int gkc = gki < K ? gki : (K - 1);
          float v = W[(size_t)gnc * K + gkc];
          lB[r][kq + i] = (gn < N && gki < K) ? f2bf(v) : (unsigned short)0;
        }
      }
    }
    // speculative prefetch of the next A K-chunk (global_prefetch_b8)
    if (kb + 32 < K) {
      int gm = m0 + r;
      int gkp = kb + 32 + kq;
      if (gm < M && gkp < K) __builtin_prefetch(A + (size_t)gm * K + gkp, 0, 1);
    }
    __syncthreads();

    const int rA  = wm * 16 + (lane & 15);
    const int cB0 = wn * 32 + (lane & 15);
    const int cB1 = cB0 + 16;
    const int ko  = (lane >> 4) * 8;      // lanes 0-15: K 0..7/16..23, lanes 16-31: +8
    v16bf a, b0, b1;
    #pragma unroll
    for (int i = 0; i < 8; ++i) {
      a[i]     = bfc(lA[rA][ko + i]);
      a[i + 8] = bfc(lA[rA][ko + 16 + i]);
      b0[i]    = bfc(lB[cB0][ko + i]);
      b0[i + 8]= bfc(lB[cB0][ko + 16 + i]);
      b1[i]    = bfc(lB[cB1][ko + i]);
      b1[i + 8]= bfc(lB[cB1][ko + 16 + i]);
    }
    acc0 = __builtin_amdgcn_wmma_f32_16x16x32_bf16(false, a, false, b0, (short)0, acc0, false, false);
    acc1 = __builtin_amdgcn_wmma_f32_16x16x32_bf16(false, a, false, b1, (short)0, acc1, false, false);
    __syncthreads();
  }

  // epilogue: C layout VGPR i <-> M = i (+8 for lanes>=16), N = lane%16
  const int rowBase = m0 + wm * 16 + ((lane >> 4) << 3);
  const int col0 = n0 + wn * 32 + (lane & 15);
  const int col1 = col0 + 16;
  float bv0 = (bias && col0 < N) ? bias[col0] : 0.0f;
  float bv1 = (bias && col1 < N) ? bias[col1] : 0.0f;
  #pragma unroll
  for (int i = 0; i < 8; ++i) {
    int row = rowBase + i;
    if (row >= M) continue;
    if (col0 < N) {
      float v = acc0[i] + bv0;
      if (act == ACT_TANH) v = tanhf(v); else if (act == ACT_RELU) v = fmaxf(v, 0.0f);
      C[(size_t)row * N + col0] = v;
    }
    if (col1 < N) {
      float v = acc1[i] + bv1;
      if (act == ACT_TANH) v = tanhf(v); else if (act == ACT_RELU) v = fmaxf(v, 0.0f);
      C[(size_t)row * N + col1] = v;
    }
  }
}

// ---------------------------------------------------------------------------
// Elementwise helpers
// ---------------------------------------------------------------------------
__global__ void k_zero(float* __restrict__ p, long n) {
  long i = (long)blockIdx.x * blockDim.x + threadIdx.x;
  if (i < n) p[i] = 0.0f;
}

// embedding gather with length masking: emb[(n*40+t)*100+e]
__global__ void k_embed(const int* __restrict__ tok, const float* __restrict__ tab,
                        const int* __restrict__ ls, float* __restrict__ emb, long total) {
  long idx = (long)blockIdx.x * blockDim.x + threadIdx.x;
  if (idx >= total) return;
  long nt = idx / 100; int e = (int)(idx % 100);
  int n = (int)(nt / 40), t = (int)(nt % 40);
  float v = 0.0f;
  if (t < ls[n]) v = tab[(size_t)tok[nt] * 100 + e];
  emb[idx] = v;
}

// One GRU cell step over Nseq sequences (H=100). dir=1 reads gi at the
// reverse-within-length index and writes the output reverse-padded.
__global__ void k_gru_update(const float* __restrict__ gi, const float* __restrict__ gh,
                             float* __restrict__ h, float* __restrict__ enc,
                             const int* __restrict__ lens,
                             int Nseq, int T, int t, int dir, int encld, int encoff) {
  int idx = blockIdx.x * blockDim.x + threadIdx.x;
  if (idx >= Nseq * 100) return;
  int n = idx / 100, j = idx % 100;
  int len = lens[n];
  int ti = t;
  if (dir) ti = (t < len) ? (len - 1 - t) : t;   // padded rows hold bias only
  const float* gir = gi + ((size_t)n * T + ti) * 300;
  const float* ghr = gh + (size_t)n * 300;
  float r  = sigmoidf_(gir[j]       + ghr[j]);
  float z  = sigmoidf_(gir[100 + j] + ghr[100 + j]);
  float nn = tanhf(gir[200 + j] + r * ghr[200 + j]);
  float hp = h[(size_t)n * 100 + j];
  float hnew = (1.0f - z) * nn + z * hp;
  h[(size_t)n * 100 + j] = hnew;
  int tw = t; bool wr = true;
  if (dir) { if (t < len) tw = len - 1 - t; else wr = false; }
  if (wr) enc[((size_t)n * T + tw) * encld + encoff + j] = hnew;
}

// EmbedAttention pooling: e = exp(score)*mask ; out = sum(e*enc)/(sum(e)+1e-4)
__global__ __launch_bounds__(128) void k_att_pool(
    const float* __restrict__ E1, const float* __restrict__ enc,
    const float* __restrict__ aw, const float* __restrict__ ab,
    const int* __restrict__ lens, float* __restrict__ out,
    int T, int D) {
  __shared__ float e[64];
  __shared__ float ssum;
  int n = blockIdx.x;
  int len = lens[n];
  for (int t = threadIdx.x; t < T; t += blockDim.x) {
    float s = ab[0];
    const float* row = E1 + ((size_t)n * T + t) * D;
    for (int d = 0; d < D; ++d) s += row[d] * aw[d];
    e[t] = (t < len) ? expf(s) : 0.0f;      // faithful: unshifted exp
  }
  __syncthreads();
  if (threadIdx.x == 0) {
    float acc = 1e-4f;
    for (int t = 0; t < T; ++t) acc += e[t];
    ssum = acc;
  }
  __syncthreads();
  float inv = 1.0f / ssum;
  for (int d = threadIdx.x; d < D; d += blockDim.x) {
    float acc = 0.0f;
    for (int t = 0; t < T; ++t) acc += e[t] * enc[((size_t)n * T + t) * D + d];
    out[(size_t)n * D + d] = acc * inv;
  }
}

// zero rows s >= ln[b] of a [B*S, D] buffer
__global__ void k_mask_rows(float* __restrict__ buf, const int* __restrict__ ln,
                            long total, int D, int S) {
  long idx = (long)blockIdx.x * blockDim.x + threadIdx.x;
  if (idx >= total) return;
  int n = (int)(idx / D);
  if ((n % S) >= ln[n / S]) buf[idx] = 0.0f;
}

// MHA core: one block per (b, q). heads=4, hd=50, Lk=20. Writes AV -> mo and
// head-averaged attention -> attn_out (the ent_attn slice of d_out).
__global__ __launch_bounds__(128) void k_mha(
    const float* __restrict__ Q, const float* __restrict__ K,
    const float* __restrict__ V, const int* __restrict__ lk,
    float* __restrict__ mo, float* __restrict__ attn_out) {
  __shared__ float sc[4][20];
  __shared__ float at[4][20];
  int row = blockIdx.x;          // b*50 + q
  int b = row / 50;
  int nk = lk[b];
  int p = threadIdx.x;
  if (p < 80) {
    int h = p / 20, k = p % 20;
    const float* qr = Q + (size_t)row * 200 + h * 50;
    const float* kr = K + ((size_t)b * 20 + k) * 200 + h * 50;
    float s = 0.0f;
    for (int d = 0; d < 50; ++d) s += qr[d] * kr[d];
    sc[h][k] = s * 0.14142135623730951f;   // 1/sqrt(50)
  }
  __syncthreads();
  if (threadIdx.x < 4) {
    int h = threadIdx.x;
    float mx = -3.4e38f;
    for (int k = 0; k < nk; ++k) mx = fmaxf(mx, sc[h][k]);
    float sum = 0.0f;
    for (int k = 0; k < 20; ++k) {
      float v = (k < nk) ? expf(sc[h][k] - mx) : 0.0f;
      at[h][k] = v; sum += v;
    }
    float inv = 1.0f / sum;
    for (int k = 0; k < 20; ++k) at[h][k] *= inv;
  }
  __syncthreads();
  for (int d = threadIdx.x; d < 200; d += blockDim.x) {
    int h = d / 50;
    float acc = 0.0f;
    for (int k = 0; k < 20; ++k) acc += at[h][k] * V[((size_t)b * 20 + k) * 200 + d];
    mo[(size_t)row * 200 + d] = acc;
  }
  if (threadIdx.x < 20) {
    int k = threadIdx.x;
    attn_out[(size_t)row * 20 + k] = 0.25f * (at[0][k] + at[1][k] + at[2][k] + at[3][k]);
  }
}

// cat[n,0:200]=sent, cat[n,200:250]=ent_out, zero where s >= ln[b]
__global__ void k_concat(const float* __restrict__ sent, const float* __restrict__ eo,
                         const int* __restrict__ ln, float* __restrict__ cat, long total) {
  long idx = (long)blockIdx.x * blockDim.x + threadIdx.x;
  if (idx >= total) return;
  int n = (int)(idx / 250), c = (int)(idx % 250);
  int b = n / 50, s = n % 50;
  float v = 0.0f;
  if (s < ln[b]) v = (c < 200) ? sent[(size_t)n * 200 + c] : eo[(size_t)n * 50 + (c - 200)];
  cat[idx] = v;
}

// ---------------------------------------------------------------------------
extern "C" void kernel_launch(void* const* d_in, const int* in_sizes, int n_in,
                              void* d_out, int out_size, void* d_ws, size_t ws_size,
                              hipStream_t stream) {
  (void)in_sizes; (void)n_in; (void)out_size; (void)ws_size;
  // ---- inputs (pytree leaf order of setup_inputs) ----
  const float* ent_embs = (const float*)d_in[0];   // [128,20,200]
  const float* emb_tab  = (const float*)d_in[1];   // [50000,100]
  const float* wif  = (const float*)d_in[2];  const float* whf  = (const float*)d_in[3];
  const float* bif  = (const float*)d_in[4];  const float* bhf  = (const float*)d_in[5];
  const float* wib  = (const float*)d_in[6];  const float* whb  = (const float*)d_in[7];
  const float* bib  = (const float*)d_in[8];  const float* bhb  = (const float*)d_in[9];
  const float* lw   = (const float*)d_in[10]; const float* lb   = (const float*)d_in[11];
  const float* aw   = (const float*)d_in[12]; const float* ab   = (const float*)d_in[13];
  const float* wifs = (const float*)d_in[14]; const float* whfs = (const float*)d_in[15];
  const float* bifs = (const float*)d_in[16]; const float* bhfs = (const float*)d_in[17];
  const float* wibs = (const float*)d_in[18]; const float* whbs = (const float*)d_in[19];
  const float* bibs = (const float*)d_in[20]; const float* bhbs = (const float*)d_in[21];
  const float* lw2  = (const float*)d_in[22]; const float* lb2  = (const float*)d_in[23];
  const float* aw2  = (const float*)d_in[24]; const float* ab2  = (const float*)d_in[25];
  const float* wq = (const float*)d_in[26]; const float* bq = (const float*)d_in[27];
  const float* wk = (const float*)d_in[28]; const float* bk = (const float*)d_in[29];
  const float* wv = (const float*)d_in[30]; const float* bv = (const float*)d_in[31];
  const float* wo = (const float*)d_in[32]; const float* bo = (const float*)d_in[33];
  const float* elw = (const float*)d_in[34]; const float* elb = (const float*)d_in[35];
  const int* tokens = (const int*)d_in[36];   // [128,50,40]
  const int* ln     = (const int*)d_in[37];   // [128]
  const int* ls     = (const int*)d_in[38];   // [128,50] -> flat [6400]
  const int* lk     = (const int*)d_in[39];   // [128]

  float* out_content = (float*)d_out;                       // [128,200]
  float* out_attn    = (float*)d_out + 128 * 200;           // [128,50,20]

  // ---- workspace partition (floats) ----
  float* p = (float*)d_ws;
  auto alloc = [&](size_t n) { float* r = p; p += n; return r; };
  const long NW = 6400;            // word-level sequences (B*S)
  const long TW_ = 40, TS = 50;
  float* emb_w = alloc((size_t)NW * TW_ * 100);
  float* gi_f  = alloc((size_t)NW * TW_ * 300);
  float* gi_b  = alloc((size_t)NW * TW_ * 300);
  float* enc_w = alloc((size_t)NW * TW_ * 200);
  float* E1    = alloc((size_t)NW * TW_ * 200);
  float* gh_f  = alloc((size_t)NW * 300);
  float* gh_b  = alloc((size_t)NW * 300);
  float* h_f   = alloc((size_t)NW * 100);
  float* h_b   = alloc((size_t)NW * 100);
  float* sent  = alloc((size_t)NW * 200);
  float* Qp    = alloc((size_t)NW * 200);
  float* Kp    = alloc((size_t)2560 * 200);
  float* Vp    = alloc((size_t)2560 * 200);
  float* mo    = alloc((size_t)NW * 200);
  float* mo2   = alloc((size_t)NW * 200);
  float* eo    = alloc((size_t)NW * 50);
  float* cat   = alloc((size_t)NW * 250);
  float* gi_sf = alloc((size_t)NW * 300);
  float* gi_sb = alloc((size_t)NW * 300);
  float* enc_s = alloc((size_t)NW * 200);
  float* E2    = alloc((size_t)NW * 200);
  float* h_sf  = alloc((size_t)128 * 100);
  float* h_sb  = alloc((size_t)128 * 100);

  auto gemm = [&](const float* A, const float* W, const float* B, float* C,
                  int M, int N, int K, int act) {
    dim3 g((N + 63) / 64, (M + 63) / 64);
    k_gemm_bf16<<<g, 256, 0, stream>>>(A, W, B, C, M, N, K, act);
  };
  auto zero = [&](float* q, long n) {
    k_zero<<<(int)((n + 255) / 256), 256, 0, stream>>>(q, n);
  };

  // ================= word level =================
  zero(enc_w, NW * TW_ * 200);
  zero(h_f, NW * 100); zero(h_b, NW * 100);
  {
    long total = NW * TW_ * 100;
    k_embed<<<(int)((total + 255) / 256), 256, 0, stream>>>(tokens, emb_tab, ls, emb_w, total);
  }
  gemm(emb_w, wif, bif, gi_f, (int)(NW * TW_), 300, 100, ACT_NONE);
  gemm(emb_w, wib, bib, gi_b, (int)(NW * TW_), 300, 100, ACT_NONE);
  for (int t = 0; t < TW_; ++t) {
    gemm(h_f, whf, bhf, gh_f, (int)NW, 300, 100, ACT_NONE);
    k_gru_update<<<(int)((NW * 100 + 255) / 256), 256, 0, stream>>>(
        gi_f, gh_f, h_f, enc_w, ls, (int)NW, (int)TW_, t, 0, 200, 0);
    gemm(h_b, whb, bhb, gh_b, (int)NW, 300, 100, ACT_NONE);
    k_gru_update<<<(int)((NW * 100 + 255) / 256), 256, 0, stream>>>(
        gi_b, gh_b, h_b, enc_w, ls, (int)NW, (int)TW_, t, 1, 200, 100);
  }
  gemm(enc_w, lw, lb, E1, (int)(NW * TW_), 200, 200, ACT_TANH);
  k_att_pool<<<(int)NW, 128, 0, stream>>>(E1, enc_w, aw, ab, ls, sent, (int)TW_, 200);
  k_mask_rows<<<(int)((NW * 200 + 255) / 256), 256, 0, stream>>>(sent, ln, NW * 200, 200, 50);

  // ================= news-entity cross attention =================
  gemm(sent, wq, bq, Qp, (int)NW, 200, 200, ACT_NONE);
  gemm(ent_embs, wk, bk, Kp, 2560, 200, 200, ACT_NONE);
  gemm(ent_embs, wv, bv, Vp, 2560, 200, 200, ACT_NONE);
  k_mha<<<(int)NW, 128, 0, stream>>>(Qp, Kp, Vp, lk, mo, out_attn);
  gemm(mo, wo, bo, mo2, (int)NW, 200, 200, ACT_NONE);
  gemm(mo2, elw, elb, eo, (int)NW, 50, 200, ACT_RELU);
  k_concat<<<(int)((NW * 250 + 255) / 256), 256, 0, stream>>>(sent, eo, ln, cat, NW * 250);

  // ================= sentence level =================
  gemm(cat, wifs, bifs, gi_sf, (int)NW, 300, 250, ACT_NONE);
  gemm(cat, wibs, bibs, gi_sb, (int)NW, 300, 250, ACT_NONE);
  zero(enc_s, NW * 200);
  zero(h_sf, 128 * 100); zero(h_sb, 128 * 100);
  for (int t = 0; t < TS; ++t) {
    gemm(h_sf, whfs, bhfs, gh_f, 128, 300, 100, ACT_NONE);
    k_gru_update<<<(128 * 100 + 255) / 256, 256, 0, stream>>>(
        gi_sf, gh_f, h_sf, enc_s, ln, 128, (int)TS, t, 0, 200, 0);
    gemm(h_sb, whbs, bhbs, gh_b, 128, 300, 100, ACT_NONE);
    k_gru_update<<<(128 * 100 + 255) / 256, 256, 0, stream>>>(
        gi_sb, gh_b, h_sb, enc_s, ln, 128, (int)TS, t, 1, 200, 100);
  }
  gemm(enc_s, lw2, lb2, E2, (int)NW, 200, 200, ACT_TANH);
  k_att_pool<<<128, 128, 0, stream>>>(E2, enc_s, aw2, ab2, ln, out_content, (int)TS, 200);
}
